// HamburgerAttentionTransformerEncoder_41085657153704
// MI455X (gfx1250) — matrix operand order repaired and checked by reading, not compiled, not run on hardware
//
#include <hip/hip_runtime.h>
#include <math.h>

// ---------------------------------------------------------------------------
// Types for CDNA5 WMMA (gfx1250, wave32).
// bf16 storage is raw u16; the only __bf16 usage is the WMMA operand vector,
// built by bit-reinterpretation of 16-byte global loads.
// ---------------------------------------------------------------------------
typedef unsigned short bf16_t;
typedef __attribute__((ext_vector_type(16))) __bf16 v16bf;
typedef __attribute__((ext_vector_type(8)))  float  v8f;

constexpr int kB = 32, kT = 1024, kF = 768, kD = 512, kR = 64, kN = 768, kH = 3072;
constexpr int kSteps = 7;

static __device__ __forceinline__ bf16_t f2bf(float f) {
  unsigned u = __float_as_uint(f);
  u += 0x7FFFu + ((u >> 16) & 1u);          // round-to-nearest-even
  return (bf16_t)(u >> 16);
}

static __device__ __forceinline__ float epilogue_act(float val, int act) {
  if (act == 1)      return fmaxf(val, 0.0f);
  else if (act == 2) return 0.5f * val * (1.0f + erff(val * 0.70710678118654752f));
  return val;
}

// ---------------------------------------------------------------------------
// Generic batched GEMM:  C[m,n] = sum_k A[m,k] * Bt[n,k]   (Bt is N x K)
// bf16 operands, fp32 WMMA accumulation (v_wmma_f32_16x16x32_bf16).
// Small-tile version: 8 waves x one 16x16 tile -> 32x64 block tile.
// Requires: M % 32 == 0, N % 64 == 0, K % 32 == 0.
// Epilogue: (+bias per n) -> act (0 none, 1 relu, 2 exact gelu) -> (+add)
//           -> optional fp32 and/or bf16 outputs.
// ---------------------------------------------------------------------------
__global__ __launch_bounds__(256) void gemm_bf16_wmma(
    const bf16_t* __restrict__ A, const bf16_t* __restrict__ Bt,
    float* __restrict__ Cf, bf16_t* __restrict__ Cb,
    const float* __restrict__ bias, const float* __restrict__ add,
    int M, int N, int K, int lda, int ldb, int ldc,
    long long sA, long long sB, long long sC, long long sAdd, int act)
{
  const int bz   = blockIdx.z;
  const int wave = threadIdx.x >> 5;
  const int lane = threadIdx.x & 31;
  const int half = lane >> 4;
  const int lr   = lane & 15;
  const int tm   = blockIdx.y * 32 + (wave >> 2) * 16;
  const int tn   = blockIdx.x * 64 + (wave & 3) * 16;

  const bf16_t* __restrict__ arow = A + (long long)bz * sA + (long long)(tm + lr) * lda;
  const bf16_t* __restrict__ brow = Bt + (long long)bz * sB + (long long)(tn + lr) * ldb;

  v8f acc = {};
  for (int k0 = 0; k0 < K; k0 += 32) {
    // A fragment (16x32 bf16): lane row = m%16; elems 0..7 <- K = k0+half*8..+7,
    // elems 8..15 <- K = k0+16+half*8..+7  (per ISA 16-bit A layout).
    union { uint4 u[2]; v16bf v; } af, bfm;
    af.u[0] = *reinterpret_cast<const uint4*>(arow + k0 + half * 8);
    af.u[1] = *reinterpret_cast<const uint4*>(arow + k0 + 16 + half * 8);
    // B fragment (32x16 bf16): lane col = n%16; same K order as A.
    bfm.u[0] = *reinterpret_cast<const uint4*>(brow + k0 + half * 8);
    bfm.u[1] = *reinterpret_cast<const uint4*>(brow + k0 + 16 + half * 8);
    acc = __builtin_amdgcn_wmma_f32_16x16x32_bf16(false, af.v, false, bfm.v,
                                                  (short)0, acc, false, false);
  }

  const int n = tn + lr;
  const float bv = bias ? bias[n] : 0.0f;
  const long long cb0 = (long long)bz * sC;
  const long long ab0 = (long long)bz * sAdd;
#pragma unroll
  for (int v = 0; v < 8; ++v) {
    const int m = tm + v + half * 8;            // C/D layout: vgpr v -> row v (+8 hi half)
    const long long ci = cb0 + (long long)m * ldc + n;
    float val = epilogue_act(acc[v] + bv, act);
    if (add) val += add[ab0 + (long long)m * ldc + n];
    if (Cf) Cf[ci] = val;
    if (Cb) Cb[ci] = f2bf(val);
  }
}

// ---------------------------------------------------------------------------
// Register-blocked GEMM for the large matmuls: each wave computes a 32x64
// tile (2 M x 4 N accumulators -> 8 WMMAs per 12 b128 loads per K-chunk).
// 8 waves arranged 4(M) x 2(N) -> 128x128 block tile.
// Requires: M % 128 == 0, N % 128 == 0, K % 32 == 0.
// ---------------------------------------------------------------------------
__global__ __launch_bounds__(256) void gemm_bf16_wmma_big(
    const bf16_t* __restrict__ A, const bf16_t* __restrict__ Bt,
    float* __restrict__ Cf, bf16_t* __restrict__ Cb,
    const float* __restrict__ bias, const float* __restrict__ add,
    int M, int N, int K, int lda, int ldb, int ldc,
    long long sA, long long sB, long long sC, long long sAdd, int act)
{
  const int bz   = blockIdx.z;
  const int wave = threadIdx.x >> 5;
  const int lane = threadIdx.x & 31;
  const int half = lane >> 4;
  const int lr   = lane & 15;
  const int tm   = blockIdx.y * 128 + (wave >> 1) * 32;   // 4 wave-rows of 32
  const int tn   = blockIdx.x * 128 + (wave & 1) * 64;    // 2 wave-cols of 64

  const bf16_t* __restrict__ a0 = A + (long long)bz * sA + (long long)(tm + lr) * lda;
  const bf16_t* __restrict__ a1 = a0 + 16LL * lda;
  const bf16_t* __restrict__ b0 = Bt + (long long)bz * sB + (long long)(tn + lr) * ldb;

  v8f acc[2][4] = {};
  for (int k0 = 0; k0 < K; k0 += 32) {
    union { uint4 u[2]; v16bf v; } af[2], bfr[4];
    af[0].u[0] = *reinterpret_cast<const uint4*>(a0 + k0 + half * 8);
    af[0].u[1] = *reinterpret_cast<const uint4*>(a0 + k0 + 16 + half * 8);
    af[1].u[0] = *reinterpret_cast<const uint4*>(a1 + k0 + half * 8);
    af[1].u[1] = *reinterpret_cast<const uint4*>(a1 + k0 + 16 + half * 8);
#pragma unroll
    for (int j = 0; j < 4; ++j) {
      const bf16_t* bj = b0 + (long long)(16 * j) * ldb;
      bfr[j].u[0] = *reinterpret_cast<const uint4*>(bj + k0 + half * 8);
      bfr[j].u[1] = *reinterpret_cast<const uint4*>(bj + k0 + 16 + half * 8);
    }
#pragma unroll
    for (int i = 0; i < 2; ++i)
#pragma unroll
      for (int j = 0; j < 4; ++j)
        acc[i][j] = __builtin_amdgcn_wmma_f32_16x16x32_bf16(
            false, af[i].v, false, bfr[j].v, (short)0, acc[i][j], false, false);
  }

  const long long cb0 = (long long)bz * sC;
  const long long ab0 = (long long)bz * sAdd;
#pragma unroll
  for (int i = 0; i < 2; ++i) {
#pragma unroll
    for (int j = 0; j < 4; ++j) {
      const int n = tn + 16 * j + lr;
      const float bv = bias ? bias[n] : 0.0f;
#pragma unroll
      for (int v = 0; v < 8; ++v) {
        const int m = tm + 16 * i + v + half * 8;
        const long long ci = cb0 + (long long)m * ldc + n;
        float val = epilogue_act(acc[i][j][v] + bv, act);
        if (add) val += add[ab0 + (long long)m * ldc + n];
        if (Cf) Cf[ci] = val;
        if (Cb) Cb[ci] = f2bf(val);
      }
    }
  }
}

// ---------------------------------------------------------------------------
// LayerNorm over F=768 (one row per 256-thread block), emits fp32 + bf16.
// ---------------------------------------------------------------------------
__global__ __launch_bounds__(256) void layernorm_kernel(
    const float* __restrict__ x, const float* __restrict__ g, const float* __restrict__ b,
    float* __restrict__ yf, bf16_t* __restrict__ yb)
{
  __shared__ float red[256];
  const long long row = blockIdx.x;
  const float* xr = x + row * kF;
  const int t = threadIdx.x;
  float v0 = xr[t], v1 = xr[t + 256], v2 = xr[t + 512];
  red[t] = v0 + v1 + v2;
  __syncthreads();
  for (int o = 128; o > 0; o >>= 1) { if (t < o) red[t] += red[t + o]; __syncthreads(); }
  const float mean = red[0] * (1.0f / kF);
  __syncthreads();
  float d0 = v0 - mean, d1 = v1 - mean, d2 = v2 - mean;
  red[t] = d0 * d0 + d1 * d1 + d2 * d2;
  __syncthreads();
  for (int o = 128; o > 0; o >>= 1) { if (t < o) red[t] += red[t + o]; __syncthreads(); }
  const float rstd = rsqrtf(red[0] * (1.0f / kF) + 1e-5f);
  float* yr = yf + row * kF;
  bf16_t* ybr = yb + row * kF;
  float o0 = d0 * rstd * g[t]       + b[t];
  float o1 = d1 * rstd * g[t + 256] + b[t + 256];
  float o2 = d2 * rstd * g[t + 512] + b[t + 512];
  yr[t] = o0; yr[t + 256] = o1; yr[t + 512] = o2;
  ybr[t] = f2bf(o0); ybr[t + 256] = f2bf(o1); ybr[t + 512] = f2bf(o2);
}

__global__ void convert_f32_bf16(const float* __restrict__ in, bf16_t* __restrict__ out,
                                 long long n)
{
  long long i = (long long)blockIdx.x * blockDim.x + threadIdx.x;
  if (i < n) out[i] = f2bf(in[i]);
}

// Tiled transpose fp32(rows,cols) -> bf16(cols,rows), batched.
__global__ __launch_bounds__(256) void transpose_f32_bf16(
    const float* __restrict__ in, bf16_t* __restrict__ out,
    int rows, int cols, long long sIn, long long sOut)
{
  __shared__ float tile[32][33];
  const float* ib = in + (long long)blockIdx.z * sIn;
  bf16_t* ob = out + (long long)blockIdx.z * sOut;
  const int c0 = blockIdx.x * 32, r0 = blockIdx.y * 32;
  const int tx = threadIdx.x & 31, ty = threadIdx.x >> 5;
#pragma unroll
  for (int i = 0; i < 32; i += 8) {
    int r = r0 + ty + i, c = c0 + tx;
    tile[ty + i][tx] = (r < rows && c < cols) ? ib[(long long)r * cols + c] : 0.0f;
  }
  __syncthreads();
#pragma unroll
  for (int i = 0; i < 32; i += 8) {
    int r = c0 + ty + i, c = r0 + tx;
    if (r < cols && c < rows) ob[(long long)r * rows + c] = f2bf(tile[tx][ty + i]);
  }
}

// L2-normalize bases over D per (b,r); emit fp32 (D,R), bf16 (D,R), bf16 (R,D).
__global__ __launch_bounds__(128) void normalize_bases_kernel(
    const float* __restrict__ braw, float* __restrict__ b32,
    bf16_t* __restrict__ bb, bf16_t* __restrict__ bbT)
{
  __shared__ float red[128];
  const int r = blockIdx.x, bz = blockIdx.y, t = threadIdx.x;
  const long long base = (long long)bz * kD * kR;
  float v[4]; float ss = 0.0f;
#pragma unroll
  for (int k = 0; k < 4; ++k) { int d = t + k * 128; v[k] = braw[base + (long long)d * kR + r]; ss += v[k] * v[k]; }
  red[t] = ss; __syncthreads();
  for (int o = 64; o > 0; o >>= 1) { if (t < o) red[t] += red[t + o]; __syncthreads(); }
  const float inv = 1.0f / fmaxf(sqrtf(red[0]), 1e-12f);
  const long long baseT = (long long)bz * kR * kD;
#pragma unroll
  for (int k = 0; k < 4; ++k) {
    int d = t + k * 128; float nv = v[k] * inv;
    b32[base + (long long)d * kR + r] = nv;
    bb [base + (long long)d * kR + r] = f2bf(nv);
    bbT[baseT + (long long)r * kD + d] = f2bf(nv);
  }
}

// Softmax over R=64 (one row per thread); emit fp32, bf16 (N,R), bf16T (R,N).
__global__ void softmax_r_kernel(const float* __restrict__ logits,
    float* __restrict__ c32, bf16_t* __restrict__ cb, bf16_t* __restrict__ cbT)
{
  const long long row = (long long)blockIdx.x * blockDim.x + threadIdx.x;
  if (row >= (long long)kB * kN) return;
  const long long bz = row / kN, n = row - bz * kN;
  const float* lr = logits + row * kR;
  float mx = -3.4e38f;
  for (int r = 0; r < kR; ++r) mx = fmaxf(mx, lr[r]);
  float s = 0.0f;
  for (int r = 0; r < kR; ++r) s += expf(lr[r] - mx);
  const float inv = 1.0f / s;
  float* o = c32 + row * kR;
  bf16_t* ob = cb + row * kR;
  bf16_t* obT = cbT + bz * (long long)kR * kN + n;
  for (int r = 0; r < kR; ++r) {
    float v = expf(lr[r] - mx) * inv;
    o[r] = v; ob[r] = f2bf(v); obT[(long long)r * kN] = f2bf(v);
  }
}

// Multiplicative update: val *= num/(den+eps); emit fp32, bf16, bf16-transposed.
__global__ void mu_update_kernel(float* __restrict__ val,
    const float* __restrict__ num, const float* __restrict__ den,
    bf16_t* __restrict__ vb, bf16_t* __restrict__ vbT, int rows, int cols)
{
  const long long i = (long long)blockIdx.x * blockDim.x + threadIdx.x;
  const long long per = (long long)rows * cols;
  if (i >= (long long)kB * per) return;
  const long long bz = i / per, rem = i - bz * per;
  const int rr = (int)(rem / cols), cc = (int)(rem - (long long)rr * cols);
  const float nv = val[i] * num[i] / (den[i] + 1e-6f);
  val[i] = nv; vb[i] = f2bf(nv);
  vbT[bz * per + (long long)cc * rows + rr] = f2bf(nv);
}

// K = relu( (up - bn_m[t])*bn_g[t]*rsqrt(bn_v[t]+eps) + bn_b[t] + h )
__global__ void bn_relu_add_kernel(const float* __restrict__ up, const float* __restrict__ h,
    const float* __restrict__ bg, const float* __restrict__ bb,
    const float* __restrict__ bm, const float* __restrict__ bv, float* __restrict__ Ko)
{
  const long long i = (long long)blockIdx.x * blockDim.x + threadIdx.x;
  if (i >= (long long)kB * kT * kF) return;
  const int t = (int)((i / kF) % kT);
  const float scale = bg[t] * rsqrtf(bv[t] + 1e-5f);
  const float v = (up[i] - bm[t]) * scale + bb[t] + h[i];
  Ko[i] = fmaxf(v, 0.0f);
}

// ysum[b,f] = sum_t softmax_t(K[b,:,f]) * V[b,t,f]   (one (b,f) per block)
__global__ __launch_bounds__(256) void attn_pool_kernel(
    const float* __restrict__ Km, const float* __restrict__ Vm, float* __restrict__ ys)
{
  __shared__ float red[256];
  const int f = blockIdx.x, bz = blockIdx.y, t = threadIdx.x;
  const float* kc = Km + (long long)bz * kT * kF + f;
  const float* vc = Vm + (long long)bz * kT * kF + f;
  float kv[4], vv[4]; float mx = -3.4e38f;
#pragma unroll
  for (int j = 0; j < 4; ++j) {
    long long tt = t + j * 256;
    kv[j] = kc[tt * kF]; vv[j] = vc[tt * kF];
    mx = fmaxf(mx, kv[j]);
  }
  red[t] = mx; __syncthreads();
  for (int o = 128; o > 0; o >>= 1) { if (t < o) red[t] = fmaxf(red[t], red[t + o]); __syncthreads(); }
  mx = red[0]; __syncthreads();
  float se = 0.0f, sev = 0.0f;
#pragma unroll
  for (int j = 0; j < 4; ++j) { float e = expf(kv[j] - mx); se += e; sev += e * vv[j]; }
  red[t] = se; __syncthreads();
  for (int o = 128; o > 0; o >>= 1) { if (t < o) red[t] += red[t + o]; __syncthreads(); }
  const float stot = red[0]; __syncthreads();
  red[t] = sev; __syncthreads();
  for (int o = 128; o > 0; o >>= 1) { if (t < o) red[t] += red[t + o]; __syncthreads(); }
  if (t == 0) ys[(long long)bz * kF + f] = red[0] / stot;
}

// Yt = sigmoid(Q) * ysum[b,f]  -> bf16 (operand of the Wo projection)
__global__ void ytq_kernel(const float* __restrict__ Q, const float* __restrict__ ys,
                           bf16_t* __restrict__ yt)
{
  const long long i = (long long)blockIdx.x * blockDim.x + threadIdx.x;
  if (i >= (long long)kB * kT * kF) return;
  const int f = (int)(i % kF);
  const long long bz = i / ((long long)kT * kF);
  const float sg = 1.0f / (1.0f + expf(-Q[i]));
  yt[i] = f2bf(sg * ys[bz * kF + f]);
}

// ---------------------------------------------------------------------------
// Host orchestration
// ---------------------------------------------------------------------------
extern "C" void kernel_launch(void* const* d_in, const int* in_sizes, int n_in,
                              void* d_out, int out_size, void* d_ws, size_t ws_size,
                              hipStream_t stream)
{
  (void)in_sizes; (void)n_in; (void)out_size; (void)ws_size;
  const float* x     = (const float*)d_in[0];
  const float* ln1_g = (const float*)d_in[1];
  const float* ln1_b = (const float*)d_in[2];
  const float* ln2_g = (const float*)d_in[3];
  const float* ln2_b = (const float*)d_in[4];
  const float* Wv_w  = (const float*)d_in[5];
  const float* Wv_bi = (const float*)d_in[6];
  const float* Wq_w  = (const float*)d_in[7];
  const float* Wq_bi = (const float*)d_in[8];
  const float* Wo_w  = (const float*)d_in[9];
  const float* Wo_bi = (const float*)d_in[10];
  const float* lw    = (const float*)d_in[11];
  const float* lb    = (const float*)d_in[12];
  const float* uw    = (const float*)d_in[13];
  const float* bn_g  = (const float*)d_in[14];
  const float* bn_b  = (const float*)d_in[15];
  const float* bn_m  = (const float*)d_in[16];
  const float* bn_v  = (const float*)d_in[17];
  const float* bases = (const float*)d_in[18];
  const float* W1    = (const float*)d_in[19];
  const float* b1    = (const float*)d_in[20];
  const float* W2    = (const float*)d_in[21];
  const float* b2    = (const float*)d_in[22];

  char* ws = (char*)d_ws; size_t off = 0;
  auto alloc = [&](size_t bytes) -> void* {
    void* p = ws + off;
    off = (off + bytes + 255) & ~(size_t)255;
    return p;
  };

  float*  h_f    = (float*) alloc(4ull * kB * kT * kF);
  bf16_t* h_b    = (bf16_t*)alloc(2ull * kB * kT * kF);
  bf16_t* hT_b   = (bf16_t*)alloc(2ull * kB * kF * kT);
  bf16_t* Wv_b16 = (bf16_t*)alloc(2ull * kF * kF);
  bf16_t* Wq_b16 = (bf16_t*)alloc(2ull * kF * kF);
  bf16_t* Wo_b16 = (bf16_t*)alloc(2ull * kF * kF);
  bf16_t* lw_b16 = (bf16_t*)alloc(2ull * kD * kT);
  bf16_t* uw_b16 = (bf16_t*)alloc(2ull * kT * kD);
  bf16_t* W1_b16 = (bf16_t*)alloc(2ull * kH * kF);
  bf16_t* W2_b16 = (bf16_t*)alloc(2ull * kF * kH);
  float*  hamT_f = (float*) alloc(4ull * kB * kN * kD);
  bf16_t* ham_b  = (bf16_t*)alloc(2ull * kB * kD * kN);
  bf16_t* hamT_b = (bf16_t*)alloc(2ull * kB * kN * kD);
  float*  bas_f  = (float*) alloc(4ull * kB * kD * kR);
  bf16_t* bas_b  = (bf16_t*)alloc(2ull * kB * kD * kR);
  bf16_t* basT_b = (bf16_t*)alloc(2ull * kB * kR * kD);
  float*  coef_f = (float*) alloc(4ull * kB * kN * kR);
  bf16_t* coef_b = (bf16_t*)alloc(2ull * kB * kN * kR);
  bf16_t* coefT_b= (bf16_t*)alloc(2ull * kB * kR * kN);
  float*  numN   = (float*) alloc(4ull * kB * kN * kR);   // also the softmax logits
  float*  denN   = (float*) alloc(4ull * kB * kN * kR);
  float*  numD   = (float*) alloc(4ull * kB * kD * kR);
  float*  denD   = (float*) alloc(4ull * kB * kD * kR);
  bf16_t* BtB_b  = (bf16_t*)alloc(2ull * kB * kR * kR);
  bf16_t* CtC_b  = (bf16_t*)alloc(2ull * kB * kR * kR);
  bf16_t* nmfT_b = (bf16_t*)alloc(2ull * kB * kN * kD);
  float*  up_f   = (float*) alloc(4ull * kB * kT * kF);
  float*  K_f    = (float*) alloc(4ull * kB * kT * kF);
  float*  V_f    = (float*) alloc(4ull * kB * kT * kF);
  float*  ysum   = (float*) alloc(4ull * kB * kF);
  bf16_t* m1_b   = (bf16_t*)alloc(2ull * kB * kT * kH);
  // Aliases (lifetimes do not overlap):
  float*  Q_f   = K_f;   // Q computed after attention pooling consumed K
  bf16_t* yt_b  = h_b;   // Yt bf16 after h_b's last use (Q projection)
  float*  out_f = up_f;  // residual "out" after up consumed
  float*  h2_f  = V_f;   // LN2 output after V consumed
  bf16_t* h2_b  = hT_b;  // LN2 bf16 after hT consumed (lower bread)

  auto gemm = [&](const bf16_t* A, const bf16_t* Bt, float* Cf, bf16_t* Cb,
                  const float* bias, const float* add,
                  int M, int N, int K, int lda, int ldb, int ldc,
                  long long sA, long long sB, long long sC, long long sAdd,
                  int batch, int act) {
    if ((M % 128 == 0) && (N % 128 == 0)) {
      dim3 g((unsigned)(N / 128), (unsigned)(M / 128), (unsigned)batch);
      gemm_bf16_wmma_big<<<g, 256, 0, stream>>>(A, Bt, Cf, Cb, bias, add, M, N, K,
                                                lda, ldb, ldc, sA, sB, sC, sAdd, act);
    } else {
      dim3 g((unsigned)(N / 64), (unsigned)(M / 32), (unsigned)batch);
      gemm_bf16_wmma<<<g, 256, 0, stream>>>(A, Bt, Cf, Cb, bias, add, M, N, K,
                                            lda, ldb, ldc, sA, sB, sC, sAdd, act);
    }
  };
  auto conv = [&](const float* in, bf16_t* out, long long n) {
    convert_f32_bf16<<<(unsigned)((n + 255) / 256), 256, 0, stream>>>(in, out, n);
  };
  const long long sNR = (long long)kN * kR, sDR = (long long)kD * kR;
  const long long sRD = (long long)kR * kD, sRN = (long long)kR * kN;
  const long long sND = (long long)kN * kD, sDN = (long long)kD * kN;
  const long long sRR = (long long)kR * kR, sTF = (long long)kT * kF;

  // 1) h = LN1(x)  (fp32 + bf16)
  layernorm_kernel<<<kB * kT, 256, 0, stream>>>(x, ln1_g, ln1_b, h_f, h_b);
  // 2) weight conversions (all consumed as the N x K "Bt" operand or M x K "A")
  conv(Wv_w, Wv_b16, (long long)kF * kF);
  conv(Wq_w, Wq_b16, (long long)kF * kF);
  conv(Wo_w, Wo_b16, (long long)kF * kF);
  conv(lw,   lw_b16, (long long)kD * kT);
  conv(uw,   uw_b16, (long long)kT * kD);
  conv(W1,   W1_b16, (long long)kH * kF);
  conv(W2,   W2_b16, (long long)kF * kH);
  // 3) hT (F x T) per batch for the lower bread
  transpose_f32_bf16<<<dim3(kF / 32, kT / 32, kB), 256, 0, stream>>>(
      h_f, hT_b, kT, kF, sTF, (long long)kF * kT);
  // 4) V = h Wv^T + bv
  gemm(h_b, Wv_b16, V_f, nullptr, Wv_bi, nullptr, kB * kT, kF, kF, kF, kF, kF,
       0, 0, 0, 0, 1, 0);
  // 5) lower bread (transposed): hamT[n,d] = relu(sum_t h[t,n] lw[d,t] + lb[d])
  gemm(hT_b, lw_b16, hamT_f, hamT_b, lb, nullptr, kN, kD, kT, kT, kT, kD,
       (long long)kF * kT, 0, sND, 0, kB, 1);
  // 6) ham (D x N) bf16 via transpose
  transpose_f32_bf16<<<dim3(kD / 32, kN / 32, kB), 256, 0, stream>>>(
      hamT_f, ham_b, kN, kD, sND, sDN);
  // 7) bases0 = L2-normalize(bases) over D
  normalize_bases_kernel<<<dim3(kR, kB), 128, 0, stream>>>(bases, bas_f, bas_b, basT_b);
  // 8) logits = x^T bases0   (N x R)
  gemm(hamT_b, basT_b, numN, nullptr, nullptr, nullptr, kN, kR, kD, kD, kD, kR,
       sND, sRD, sNR, 0, kB, 0);
  // 9) coef = softmax_R(logits)
  softmax_r_kernel<<<(kB * kN + 255) / 256, 256, 0, stream>>>(numN, coef_f, coef_b, coefT_b);
  // 10) 7 multiplicative-update steps + 1 differentiable coef half-step
  for (int s = 0; s < kSteps + 1; ++s) {
    gemm(hamT_b, basT_b, numN, nullptr, nullptr, nullptr, kN, kR, kD, kD, kD, kR,
         sND, sRD, sNR, 0, kB, 0);                                    // num  = x^T B
    gemm(basT_b, basT_b, nullptr, BtB_b, nullptr, nullptr, kR, kR, kD, kD, kD, kR,
         sRD, sRD, sRR, 0, kB, 0);                                    // BtB (symmetric)
    gemm(coef_b, BtB_b, denN, nullptr, nullptr, nullptr, kN, kR, kR, kR, kR, kR,
         sNR, sRR, sNR, 0, kB, 0);                                    // C BtB
    mu_update_kernel<<<(kB * kN * kR + 255) / 256, 256, 0, stream>>>(
        coef_f, numN, denN, coef_b, coefT_b, kN, kR);                 // coef update
    if (s == kSteps) break;                                           // final half-step
    gemm(ham_b, coefT_b, numD, nullptr, nullptr, nullptr, kD, kR, kN, kN, kN, kR,
         sDN, sRN, sDR, 0, kB, 0);                                    // num2 = x C
    gemm(coefT_b, coefT_b, nullptr, CtC_b, nullptr, nullptr, kR, kR, kN, kN, kN, kR,
         sRN, sRN, sRR, 0, kB, 0);                                    // CtC (symmetric)
    gemm(bas_b, CtC_b, denD, nullptr, nullptr, nullptr, kD, kR, kR, kR, kR, kR,
         sDR, sRR, sDR, 0, kB, 0);                                    // B CtC
    mu_update_kernel<<<(kB * kD * kR + 255) / 256, 256, 0, stream>>>(
        bas_f, numD, denD, bas_b, basT_b, kD, kR);                    // bases update
  }
  // 11) nmfT[n,d] = sum_r coef[n,r] bases[d,r]   (bf16 only, feeds upper bread)
  gemm(coef_b, bas_b, nullptr, nmfT_b, nullptr, nullptr, kN, kD, kR, kR, kR, kD,
       sNR, sDR, sND, 0, kB, 0);
  // 12) upper bread: up[t,n] = sum_d uw[t,d] nmf[d,n]
  gemm(uw_b16, nmfT_b, up_f, nullptr, nullptr, nullptr, kT, kN, kD, kD, kD, kN,
       0, sND, sTF, 0, kB, 0);
  // 13) K = relu(BN_eval(up) + h)
  bn_relu_add_kernel<<<(kB * kT * kF + 255) / 256, 256, 0, stream>>>(
      up_f, h_f, bn_g, bn_b, bn_m, bn_v, K_f);
  // 14) ysum[b,f] = sum_t softmax_T(K) * V
  attn_pool_kernel<<<dim3(kF, kB), 256, 0, stream>>>(K_f, V_f, ysum);
  // 15) Q = h Wq^T + bq   (reuses K buffer)
  gemm(h_b, Wq_b16, Q_f, nullptr, Wq_bi, nullptr, kB * kT, kF, kF, kF, kF, kF,
       0, 0, 0, 0, 1, 0);
  // 16) Yt = sigmoid(Q) * ysum  (bf16, reuses h_b)
  ytq_kernel<<<(kB * kT * kF + 255) / 256, 256, 0, stream>>>(Q_f, ysum, yt_b);
  // 17) out = Yt Wo^T + bo + x   (reuses up buffer)
  gemm(yt_b, Wo_b16, out_f, nullptr, Wo_bi, x, kB * kT, kF, kF, kF, kF, kF,
       0, 0, 0, 0, 1, 0);
  // 18) h2 = LN2(out)
  layernorm_kernel<<<kB * kT, 256, 0, stream>>>(out_f, ln2_g, ln2_b, h2_f, h2_b);
  // 19) m1 = gelu(h2 W1^T + b1)   (bf16 only)
  gemm(h2_b, W1_b16, nullptr, m1_b, b1, nullptr, kB * kT, kH, kF, kF, kF, kH,
       0, 0, 0, 0, 1, 2);
  // 20) d_out = gelu(m1 W2^T + b2) + out
  gemm(m1_b, W2_b16, (float*)d_out, nullptr, b2, out_f, kB * kT, kF, kH, kH, kH, kF,
       0, 0, 0, 0, 1, 2);
}
